// Allegro_56109452755157
// MI455X (gfx1250) — compile-verified, matrix operand order checked by reading, not run on hardware
//
#include <hip/hip_runtime.h>
#include <hip/hip_bf16.h>

typedef __attribute__((ext_vector_type(16))) _Float16 v16h;
typedef __attribute__((ext_vector_type(8)))  _Float16 v8h;
typedef __attribute__((ext_vector_type(8)))  float    v8f;

#define WAVES_PER_BLOCK 4
#define BLOCK_THREADS   128

// ---- weight arena fragment offsets (in halfs). One fragment = 512 halfs (32 lanes x 16). ----
#define OFF_E0   0        // 96x64   -> 12 frags
#define OFF_E1   6144     // 64x128  -> 16
#define OFF_E2   14336    // 128x256 -> 64
#define OFF_E3   47104    // 256x64  -> 32
#define OFF_W0   63488    // 64x32   -> 4
#define OFF_W1   65536
#define OFF_M00  67584    // 96x64   -> 12
#define OFF_M01  73728
#define OFF_M10  79872    // 64x64   -> 8
#define OFF_M11  83968
#define OFF_M20  88064
#define OFF_M21  92160
#define OFF_R0   96256    // 64x64   -> 8
#define ARENA_HALFS 100352

union V16U { v16h v; v8h h[2]; };

__device__ inline v8f wmma_f16(v16h a, v16h b, v8f c) {
  return __builtin_amdgcn_wmma_f32_16x16x32_f16(false, a, false, b, (short)0, c, false, false);
}

// Load a 16x32 A-fragment from a row-major f16 tile (ld halfs, 16B-aligned rows).
// ISA A-layout: lane<16 -> M=lane, K = {0..7,16..23}; lane>=16 -> M=lane-16, K = {8..15,24..31}.
__device__ inline v16h load_a_rm(const _Float16* tile, int ld, int lane) {
  int m  = lane & 15;
  int ks = (lane & 16) ? 8 : 0;
  const _Float16* p = tile + m * ld + ks;
  V16U u;
  u.h[0] = *(const v8h*)(p);
  u.h[1] = *(const v8h*)(p + 16);
  return u.v;
}

// One dense layer on a 16-row tile living in LDS staging (row-major, leading dim ld).
// B fragments are read from the pre-packed global arena. In-place safe.
template <int KPAD, int NOUT, bool ACT>
__device__ inline void mlp_layer(_Float16* stag, int ld, const _Float16* wf,
                                 float scale, const float* env16, int lane) {
  constexpr int KT = KPAD / 32, NT = NOUT / 16;
  v8f acc[NT] = {};
#pragma unroll
  for (int kt = 0; kt < KT; ++kt) {
    v16h a = load_a_rm(stag + kt * 32, ld, lane);
#pragma unroll
    for (int nt = 0; nt < NT; ++nt) {
      v16h b = *(const v16h*)(wf + (size_t)(kt * NT + nt) * 512 + lane * 16);
      acc[nt] = wmma_f16(a, b, acc[nt]);
    }
  }
  int n = lane & 15, mb = (lane & 16) ? 8 : 0;
#pragma unroll
  for (int nt = 0; nt < NT; ++nt)
#pragma unroll
    for (int r = 0; r < 8; ++r) {
      float v = acc[nt][r] * scale;
      if (ACT) v = v / (1.f + __expf(-v));
      if (env16) v *= env16[mb + r];
      stag[(mb + r) * ld + nt * 16 + n] = (_Float16)v;
    }
}

// ---- prep: f32 weight (Ktrue x N, row-major) -> f16 B-fragment layout (A-layout of B^T) ----
__global__ void prep_weight(const float* __restrict__ src, _Float16* __restrict__ dst,
                            int Ktrue, int Kpad, int N) {
  int total = (Kpad / 32) * (N / 16) * 512;
  int NT = N / 16;
  for (int i = blockIdx.x * blockDim.x + threadIdx.x; i < total; i += gridDim.x * blockDim.x) {
    int f = i >> 9, r = i & 511;
    int lane = r >> 4, e = r & 15;
    int kt = f / NT, nt = f % NT;
    int n = nt * 16 + (lane & 15);
    int khalf = (lane & 16) ? 8 : 0;
    int k = kt * 32 + khalf + ((e < 8) ? e : (e + 8));
    float v = (k < Ktrue) ? src[(size_t)k * N + n] : 0.f;
    dst[i] = (_Float16)v;
  }
}

__global__ void zero_f32(float* __restrict__ p, int n) {
  int i = blockIdx.x * blockDim.x + threadIdx.x;
  if (i < n) p[i] = 0.f;
}

__global__ void node_init(float* __restrict__ node_e, const int* __restrict__ species,
                          const float* __restrict__ pe, int n) {
  int i = blockIdx.x * blockDim.x + threadIdx.x;
  if (i < n) node_e[i] = pe[species[i]];
}

// ---- K1: featurize + embedding MLP (72->64->128->256->64) + wv/V init ----
__global__ __launch_bounds__(BLOCK_THREADS) void k1_edge(
    const float* __restrict__ vectors, const int* __restrict__ senders,
    const int* __restrict__ receivers, const int* __restrict__ species,
    const float* __restrict__ emb, const _Float16* __restrict__ arena,
    const float* __restrict__ Wwvec, const float* __restrict__ Wvinit,
    _Float16* __restrict__ xf, float* __restrict__ envg, float* __restrict__ Yg,
    float* __restrict__ Vg, int E) {
  __shared__ _Float16 stag_s[WAVES_PER_BLOCK][16 * 256];
  __shared__ float env_s[WAVES_PER_BLOCK][16];
  __shared__ float y1_s[WAVES_PER_BLOCK][16][3];
  int wave = threadIdx.x >> 5, lane = threadIdx.x & 31;
  int tile = blockIdx.x * WAVES_PER_BLOCK + wave;
  if (tile * 16 >= E) return;
  _Float16* stag = stag_s[wave];
  float* env16 = env_s[wave];

  if (lane < 16) {
    int e = tile * 16 + lane; if (e >= E) e = E - 1;
    float vx = vectors[e * 3 + 0], vy = vectors[e * 3 + 1], vz = vectors[e * 3 + 2];
    float d = fmaxf(sqrtf(vx * vx + vy * vy + vz * vz), 1e-6f);
    float d2 = d * d, d3 = d2 * d, d6 = d3 * d3, d7 = d6 * d, d8 = d7 * d;
    float env = (d < 1.f) ? (1.f - 28.f * d6 + 48.f * d7 - 21.f * d8) : 0.f;
    env16[lane] = env;
    envg[e] = env;
    float inv_d = 1.f / d;
    const float SQ3 = 1.7320508075688772f;
    float y1x = SQ3 * vx * inv_d, y1y = SQ3 * vy * inv_d, y1z = SQ3 * vz * inv_d;
    y1_s[wave][lane][0] = y1x; y1_s[wave][lane][1] = y1y; y1_s[wave][lane][2] = y1z;
    Yg[e * 4 + 0] = 1.f; Yg[e * 4 + 1] = y1x; Yg[e * 4 + 2] = y1y; Yg[e * 4 + 3] = y1z;
    const float SQ2 = 1.4142135623730951f, PI = 3.14159265358979323846f;
    _Float16* row = stag + lane * 256;
#pragma unroll
    for (int n = 1; n <= 8; ++n) row[n - 1] = (_Float16)(SQ2 * __sinf((float)n * PI * d) * inv_d * env);
    int zs = species[senders[e]], zr = species[receivers[e]];
#pragma unroll
    for (int k = 0; k < 32; ++k) row[8 + k]  = (_Float16)emb[zs * 32 + k];
#pragma unroll
    for (int k = 0; k < 32; ++k) row[40 + k] = (_Float16)emb[zr * 32 + k];
#pragma unroll
    for (int k = 72; k < 96; ++k) row[k] = (_Float16)0.f;
  }
  __builtin_amdgcn_wave_barrier();

  mlp_layer<96, 64, true>(stag, 256, arena + OFF_E0, 0.11785113019775793f /*1/sqrt(72)*/, nullptr, lane);
  mlp_layer<64, 128, true>(stag, 256, arena + OFF_E1, 0.125f, nullptr, lane);
  mlp_layer<128, 256, true>(stag, 256, arena + OFF_E2, 0.08838834764831845f /*1/sqrt(128)*/, nullptr, lane);
  mlp_layer<256, 64, false>(stag, 256, arena + OFF_E3, 0.0625f, env16, lane);
  __builtin_amdgcn_wave_barrier();

  // write x (env-scaled, f16) to global: 2 lanes per row
  {
    int m = lane >> 1, half = lane & 1;
    int e = tile * 16 + m;
    const _Float16* srcp = stag + m * 256 + half * 32;
    _Float16* dstp = xf + (size_t)e * 64 + half * 32;
#pragma unroll
    for (int i = 0; i < 4; ++i) *(v8h*)(dstp + i * 8) = *(const v8h*)(srcp + i * 8);
  }
  // wv = x @ W_wvec / 8 ; V = wv/66 * W_vinit[c] * Y1
  if (lane < 16) {
    int e = tile * 16 + lane;
    const _Float16* row = stag + lane * 256;
    float wv = 0.f;
#pragma unroll
    for (int k = 0; k < 64; ++k) wv += (float)row[k] * Wwvec[k];
    wv *= 0.125f;
    float vs = wv / 66.f;  // n_irreps = 2 + 2*32
    float y1x = y1_s[wave][lane][0], y1y = y1_s[wave][lane][1], y1z = y1_s[wave][lane][2];
    float* vp = Vg + (size_t)e * 96;
#pragma unroll
    for (int c = 0; c < 32; ++c) {
      float coef = vs * Wvinit[c];
      vp[c * 3 + 0] = coef * y1x; vp[c * 3 + 1] = coef * y1y; vp[c * 3 + 2] = coef * y1z;
    }
  }
}

// ---- K2a: w = x @ W_w[l] / 8 ; scatter w*Y into node accumulator (senders) ----
__global__ __launch_bounds__(BLOCK_THREADS) void k2a(
    const _Float16* __restrict__ xf, const _Float16* __restrict__ wfrag,
    const int* __restrict__ senders, const float* __restrict__ Yg,
    float* __restrict__ wYnode, int E) {
  int wave = threadIdx.x >> 5, lane = threadIdx.x & 31;
  int tile = blockIdx.x * WAVES_PER_BLOCK + wave;
  if (tile * 16 >= E) return;
  v8f acc[2] = {};
  const _Float16* a0 = xf + (size_t)tile * 16 * 64;
#pragma unroll
  for (int kt = 0; kt < 2; ++kt) {
    v16h a = load_a_rm(a0 + kt * 32, 64, lane);
#pragma unroll
    for (int nt = 0; nt < 2; ++nt) {
      v16h b = *(const v16h*)(wfrag + (size_t)(kt * 2 + nt) * 512 + lane * 16);
      acc[nt] = wmma_f16(a, b, acc[nt]);
    }
  }
  int n = lane & 15, mb = (lane & 16) ? 8 : 0;
#pragma unroll
  for (int nt = 0; nt < 2; ++nt)
#pragma unroll
    for (int r = 0; r < 8; ++r) {
      int m = mb + r, e = tile * 16 + m;
      float w = acc[nt][r] * 0.125f;
      int c = nt * 16 + n;
      int s = senders[e];
      float* base = wYnode + (size_t)s * 128 + c * 4;
#pragma unroll
      for (int d = 0; d < 4; ++d) unsafeAtomicAdd(base + d, w * Yg[e * 4 + d]);
    }
}

// ---- K2b: gather wY, equivariant contraction, message MLP (96->64->64->64), W_V mix ----
__global__ __launch_bounds__(BLOCK_THREADS) void k2b(
    _Float16* __restrict__ xf, const float* __restrict__ wYnode,
    const int* __restrict__ senders, float* __restrict__ Vg,
    const float* __restrict__ WV, const _Float16* __restrict__ arena,
    int off_m0, int off_m1, int off_m2,
    const float* __restrict__ envg, const float* __restrict__ vareps, int E) {
  __shared__ _Float16 stag_s[WAVES_PER_BLOCK][16 * 96];
  __shared__ float vout_s[WAVES_PER_BLOCK][16][96];
  __shared__ float env_s[WAVES_PER_BLOCK][16];
  int wave = threadIdx.x >> 5, lane = threadIdx.x & 31;
  int tile = blockIdx.x * WAVES_PER_BLOCK + wave;
  if (tile * 16 >= E) return;
  _Float16* stag = stag_s[wave];
  float* env16 = env_s[wave];
  float eps = rsqrtf(1.f + logf(1.f + expf(vareps[0])));

  if (lane < 16) {
    int e = tile * 16 + lane;
    env16[lane] = envg[e];
    _Float16* row = stag + lane * 96;
    const _Float16* xr = xf + (size_t)e * 64;
#pragma unroll
    for (int i = 0; i < 8; ++i) *(v8h*)(row + i * 8) = *(const v8h*)(xr + i * 8);
    int s = senders[e];
    const float* wYr = wYnode + (size_t)s * 128;
    const float* Vr = Vg + (size_t)e * 96;
    float* vo = vout_s[wave][lane];
    const float r3 = 0.5773502691896258f;  // 1/sqrt(3)
#pragma unroll
    for (int c = 0; c < 32; ++c) {
      float a  = wYr[c * 4 + 0] * eps;
      float A0 = wYr[c * 4 + 1] * eps, A1 = wYr[c * 4 + 2] * eps, A2 = wYr[c * 4 + 3] * eps;
      float V0 = Vr[c * 3 + 0], V1 = Vr[c * 3 + 1], V2 = Vr[c * 3 + 2];
      row[64 + c] = (_Float16)((A0 * V0 + A1 * V1 + A2 * V2) * r3);
      vo[c * 3 + 0] = a * V0; vo[c * 3 + 1] = a * V1; vo[c * 3 + 2] = a * V2;
    }
    const float rM = 0.17677669529663687f;  // 1/sqrt(32)
    float* Vw = Vg + (size_t)e * 96;
    for (int f = 0; f < 32; ++f) {
      float s0 = 0.f, s1 = 0.f, s2 = 0.f;
#pragma unroll
      for (int c = 0; c < 32; ++c) {
        float w = WV[c * 32 + f];
        s0 += vo[c * 3 + 0] * w; s1 += vo[c * 3 + 1] * w; s2 += vo[c * 3 + 2] * w;
      }
      Vw[f * 3 + 0] = s0 * rM; Vw[f * 3 + 1] = s1 * rM; Vw[f * 3 + 2] = s2 * rM;
    }
  }
  __builtin_amdgcn_wave_barrier();

  mlp_layer<96, 64, true>(stag, 96, arena + off_m0, 0.10206207261596575f /*1/sqrt(96)*/, nullptr, lane);
  mlp_layer<64, 64, true>(stag, 96, arena + off_m1, 0.125f, nullptr, lane);
  mlp_layer<64, 64, false>(stag, 96, arena + off_m2, 0.125f, env16, lane);
  __builtin_amdgcn_wave_barrier();

  {
    int m = lane >> 1, half = lane & 1;
    int e = tile * 16 + m;
    _Float16* dst = xf + (size_t)e * 64 + half * 32;
    const _Float16* src = stag + m * 96 + half * 32;
#pragma unroll
    for (int i = 0; i < 4; ++i) *(v8h*)(dst + i * 8) = *(const v8h*)(src + i * 8);
  }
}

// ---- K3: readout x@W_r0/8, dot W_rout/8, env scale, atomic scatter to receivers ----
__global__ __launch_bounds__(BLOCK_THREADS) void k3(
    const _Float16* __restrict__ xf, const _Float16* __restrict__ wfrag,
    const float* __restrict__ Wrout, const int* __restrict__ receivers,
    const float* __restrict__ envg, float* __restrict__ node_e, int E) {
  __shared__ _Float16 stag_s[WAVES_PER_BLOCK][16 * 64];
  int wave = threadIdx.x >> 5, lane = threadIdx.x & 31;
  int tile = blockIdx.x * WAVES_PER_BLOCK + wave;
  if (tile * 16 >= E) return;
  _Float16* stag = stag_s[wave];
  v8f acc[4] = {};
  const _Float16* a0 = xf + (size_t)tile * 16 * 64;
#pragma unroll
  for (int kt = 0; kt < 2; ++kt) {
    v16h a = load_a_rm(a0 + kt * 32, 64, lane);
#pragma unroll
    for (int nt = 0; nt < 4; ++nt) {
      v16h b = *(const v16h*)(wfrag + (size_t)(kt * 4 + nt) * 512 + lane * 16);
      acc[nt] = wmma_f16(a, b, acc[nt]);
    }
  }
  int n = lane & 15, mb = (lane & 16) ? 8 : 0;
#pragma unroll
  for (int nt = 0; nt < 4; ++nt)
#pragma unroll
    for (int r = 0; r < 8; ++r)
      stag[(mb + r) * 64 + nt * 16 + n] = (_Float16)(acc[nt][r] * 0.125f);
  __builtin_amdgcn_wave_barrier();
  if (lane < 16) {
    int e = tile * 16 + lane;
    const _Float16* row = stag + lane * 64;
    float s = 0.f;
#pragma unroll
    for (int k = 0; k < 64; ++k) s += (float)row[k] * Wrout[k];
    s *= 0.125f * envg[e];
    unsafeAtomicAdd(&node_e[receivers[e]], s);
  }
}

static inline size_t align_up(size_t x, size_t a) { return (x + a - 1) / a * a; }

extern "C" void kernel_launch(void* const* d_in, const int* in_sizes, int n_in,
                              void* d_out, int out_size, void* d_ws, size_t ws_size,
                              hipStream_t stream) {
  const float* vectors  = (const float*)d_in[0];
  const int*   senders  = (const int*)d_in[1];
  const int*   receivers= (const int*)d_in[2];
  const int*   species  = (const int*)d_in[3];
  const float* emb      = (const float*)d_in[4];
  const float* W_e0     = (const float*)d_in[5];
  const float* W_e1     = (const float*)d_in[6];
  const float* W_e2     = (const float*)d_in[7];
  const float* W_e3     = (const float*)d_in[8];
  const float* W_wvec   = (const float*)d_in[9];
  const float* W_vinit  = (const float*)d_in[10];
  const float* W_w      = (const float*)d_in[11];
  const float* W_m0     = (const float*)d_in[12];
  const float* W_m1     = (const float*)d_in[13];
  const float* W_m2     = (const float*)d_in[14];
  const float* W_V      = (const float*)d_in[15];
  const float* W_r0     = (const float*)d_in[16];
  const float* W_rout   = (const float*)d_in[17];
  const float* pe       = (const float*)d_in[18];
  const float* vareps   = (const float*)d_in[19];
  int E  = in_sizes[1];
  int NN = in_sizes[3];
  if (E <= 0 || NN <= 0) return;

  char* ws = (char*)d_ws;
  size_t off = 0;
  _Float16* arena = (_Float16*)(ws + off); off = align_up(off + (size_t)ARENA_HALFS * 2, 256);
  _Float16* xf    = (_Float16*)(ws + off); off = align_up(off + (size_t)E * 64 * 2, 256);
  float* envg     = (float*)(ws + off);    off = align_up(off + (size_t)E * 4, 256);
  float* Yg       = (float*)(ws + off);    off = align_up(off + (size_t)E * 16, 256);
  float* Vg       = (float*)(ws + off);    off = align_up(off + (size_t)E * 96 * 4, 256);
  float* wYnode   = (float*)(ws + off);    off = align_up(off + (size_t)NN * 128 * 4, 256);
  (void)ws_size; (void)n_in; (void)out_size;

  auto prep = [&](const float* src, int Kt, int Kp, int N, int offh) {
    int total = (Kp / 32) * (N / 16) * 512;
    prep_weight<<<(total + 255) / 256, 256, 0, stream>>>(src, arena + offh, Kt, Kp, N);
  };
  prep(W_e0, 72, 96, 64,  OFF_E0);
  prep(W_e1, 64, 64, 128, OFF_E1);
  prep(W_e2, 128, 128, 256, OFF_E2);
  prep(W_e3, 256, 256, 64, OFF_E3);
  prep(W_w,             64, 64, 32, OFF_W0);
  prep(W_w + 64 * 32,   64, 64, 32, OFF_W1);
  prep(W_m0,            96, 96, 64, OFF_M00);
  prep(W_m0 + 96 * 64,  96, 96, 64, OFF_M01);
  prep(W_m1,            64, 64, 64, OFF_M10);
  prep(W_m1 + 64 * 64,  64, 64, 64, OFF_M11);
  prep(W_m2,            64, 64, 64, OFF_M20);
  prep(W_m2 + 64 * 64,  64, 64, 64, OFF_M21);
  prep(W_r0,            64, 64, 64, OFF_R0);

  int tiles = (E + 15) / 16;
  int blocks = (tiles + WAVES_PER_BLOCK - 1) / WAVES_PER_BLOCK;

  k1_edge<<<blocks, BLOCK_THREADS, 0, stream>>>(vectors, senders, receivers, species, emb,
                                                arena, W_wvec, W_vinit, xf, envg, Yg, Vg, E);
  for (int l = 0; l < 2; ++l) {
    int zn = NN * 128;
    zero_f32<<<(zn + 255) / 256, 256, 0, stream>>>(wYnode, zn);
    k2a<<<blocks, BLOCK_THREADS, 0, stream>>>(xf, arena + (l ? OFF_W1 : OFF_W0), senders, Yg, wYnode, E);
    k2b<<<blocks, BLOCK_THREADS, 0, stream>>>(xf, wYnode, senders, Vg, W_V + (size_t)l * 32 * 32,
                                              arena, l ? OFF_M01 : OFF_M00, l ? OFF_M11 : OFF_M10,
                                              l ? OFF_M21 : OFF_M20, envg, vareps, E);
  }
  node_init<<<(NN + 255) / 256, 256, 0, stream>>>((float*)d_out, species, pe, NN);
  k3<<<blocks, BLOCK_THREADS, 0, stream>>>(xf, arena + OFF_R0, W_rout, receivers, envg, (float*)d_out, E);
}
